// CVQVAE_20289425507063
// MI455X (gfx1250) — matrix operand order repaired and checked by reading, not compiled, not run on hardware
//
#include <hip/hip_runtime.h>

// MI455X / gfx1250 VQ-VAE forward.
// Convs: implicit GEMM on v_wmma_f32_16x16x32_f16 (wave32), fp32 accumulate,
//        f16 activations. Input patches staged WGP-locally via
//        global_load_async_to_lds_b32 (ASYNCcnt) into LDS; B (im2col)
//        fragments built from LDS, A (weight) fragments pre-packed into the
//        ISA 7.12.2 layout and reused across two output rows per load.
// VQ:    distance GEMM on WMMA (argmin_n ||e_n||^2 - 2 z.e_n), 16-lane
//        xor-shuffle arg-min.

typedef __attribute__((ext_vector_type(16))) _Float16 v16h;
typedef __attribute__((ext_vector_type(8)))  float    v8f;

__device__ __forceinline__ float selu_f(float x) {
  const float scale = 1.0507009873554805f;
  const float alpha = 1.6732632423543772f;
  return x > 0.f ? scale * x : scale * alpha * (__expf(x) - 1.f);
}

// Async global->LDS copy of 4 bytes (CDNA5 GLOBAL_LOAD_ASYNC_TO_LDS_B32).
__device__ __forceinline__ void async_g2l_b32(uint32_t lds_byte_off, uint64_t gaddr) {
  asm volatile("global_load_async_to_lds_b32 %0, %1, off"
               :: "v"(lds_byte_off), "v"(gaddr) : "memory");
}
__device__ __forceinline__ void wait_async0() {
  asm volatile("s_wait_asynccnt 0x0" ::: "memory");
}

// ---------------------------------------------------------------- prep kernels

__global__ void k_f32_to_f16(const float* __restrict__ in,
                             _Float16* __restrict__ out, int n) {
  int i = blockIdx.x * blockDim.x + threadIdx.x;
  if (i < n) out[i] = (_Float16)in[i];
}

// Pack conv weights into WMMA A-fragment layout:
//   out[((mt*Ksteps + ks)*32 + lane)*16 + j]
// A 16x32 f16: lane l holds M=l&15; K = ks*32 + (j<8 ? hi*8+j : 16+hi*8+(j-8))
// ioLayout==0: w is (Cout,Cin,5,5); ioLayout==1: w is (Cin,Cout,5,5) (conv_transpose,
// no spatial flip: jax.lax.conv_transpose default transpose_kernel=False).
__global__ void k_pack_w(const float* __restrict__ w, _Float16* __restrict__ out,
                         int Cout, int Cin, int Ksteps, int ioLayout, int total) {
  int tid = blockIdx.x * blockDim.x + threadIdx.x;
  if (tid >= total) return;
  int j    = tid & 15;
  int lane = (tid >> 4) & 31;
  int rest = tid >> 9;
  int ks   = rest % Ksteps;
  int mt   = rest / Ksteps;
  int M    = mt * 16 + (lane & 15);
  int hi   = lane >> 4;
  int K    = ks * 32 + (j < 8 ? hi * 8 + j : 16 + hi * 8 + (j - 8));
  float v = 0.f;
  if (M < Cout && K < Cin * 25) {
    int ci = K / 25, rem = K % 25;
    int ky = rem / 5, kx = rem % 5;
    if (ioLayout) v = w[(ci * Cout + M) * 25 + ky * 5 + kx];
    else          v = w[(M * Cin + ci) * 25 + ky * 5 + kx];
  }
  out[tid] = (_Float16)v;
}

// Pack codebook (512x64 f32) into WMMA B-fragment layout:
//   efrag[((et*2 + ks)*32 + lane)*16 + j];  lane l: N=l&15, K=ks*32+hi*16+j
__global__ void k_pack_emb(const float* __restrict__ emb,
                           _Float16* __restrict__ efrag) {
  int tid  = blockIdx.x * blockDim.x + threadIdx.x;  // 32768 total
  int j    = tid & 15;
  int lane = (tid >> 4) & 31;
  int ks   = (tid >> 9) & 1;
  int et   = tid >> 10;
  int N    = et * 16 + (lane & 15);
  int K    = ks * 32 + (lane >> 4) * 16 + j;
  efrag[tid] = (_Float16)emb[N * 64 + K];
}

__global__ void k_emb_norm(const float* __restrict__ emb, float* __restrict__ e2) {
  int n = blockIdx.x * blockDim.x + threadIdx.x;
  if (n < 512) {
    float s = 0.f;
    for (int d = 0; d < 64; ++d) { float v = emb[n * 64 + d]; s += v * v; }
    e2[n] = s;
  }
}

// ------------------------------------------- LDS-staged implicit-GEMM conv
// Block = 256 threads = 8 waves, one 16x16 output tile (b, ty, tx).
// Stage Cin x 20 x 20 input patch (padding resolved) into LDS via async copy.
// Wave w computes output rows {2w, 2w+1}: per (mtile, kstep) one A-fragment
// load feeds two WMMAs (two B fragments gathered from LDS).
__global__ void __launch_bounds__(256) k_conv_wmma(
    const _Float16* __restrict__ in, const _Float16* __restrict__ wfrag,
    const float* __restrict__ bias,
    _Float16* __restrict__ outh, float* __restrict__ outf,
    int Cin, int Cout, int Hin, int Win, int Hout, int Wout,
    int pad, int Ksteps, int Mtiles, int nty, int ntx, int doSelu) {
  extern __shared__ _Float16 sh[];  // Cin * 20 * 20 halfs
  const int lane = threadIdx.x & 31;
  const int wave = threadIdx.x >> 5;
  const int col  = lane & 15;
  const int hi   = lane >> 4;

  const int tpb = nty * ntx;
  const int b   = blockIdx.x / tpb;
  const int rem = blockIdx.x % tpb;
  const int oy0 = (rem / ntx) * 16;
  const int ox0 = (rem % ntx) * 16;

  // ---- stage patch: Cin*20*20 halfs as Cin*200 dword pairs
  const int npair = Cin * 200;
  for (int e = threadIdx.x; e < npair; e += 256) {
    int ci  = e / 200;
    int rr  = e - ci * 200;
    int py  = rr / 10;
    int px  = (rr - py * 10) * 2;
    int iy  = oy0 + py - pad;
    int ix  = ox0 + px - pad;
    int si  = (ci * 20 + py) * 20 + px;
    bool vy = (iy >= 0 && iy < Hin);
    bool v0 = vy && (ix >= 0 && ix < Win);
    bool v1 = vy && (ix + 1 >= 0 && ix + 1 < Win);
    if (v0 && v1) {
      uint64_t ga = (uint64_t)(uintptr_t)&in[((b * Cin + ci) * Hin + iy) * Win + ix];
      uint32_t lo = (uint32_t)(uintptr_t)&sh[si];
      async_g2l_b32(lo, ga);
    } else {
      sh[si]     = v0 ? in[((b * Cin + ci) * Hin + iy) * Win + ix]     : (_Float16)0.f;
      sh[si + 1] = v1 ? in[((b * Cin + ci) * Hin + iy) * Win + ix + 1] : (_Float16)0.f;
    }
  }
  wait_async0();
  __syncthreads();

  const int ly0 = 2 * wave;             // two output rows per wave
  const int ox  = ox0 + col;

  for (int mt = 0; mt < Mtiles; ++mt) {
    const _Float16* wf = wfrag + (size_t)mt * Ksteps * 512 + lane * 16;
    v8f acc0 = {}, acc1 = {};
    for (int ks = 0; ks < Ksteps; ++ks) {
      v16h a = *(const v16h*)(wf + (size_t)ks * 512);
      const int kbase = ks * 32 + hi * 8;
      v16h b0, b1;
#pragma unroll
      for (int j = 0; j < 16; ++j) {
        int K  = kbase + (j < 8 ? j : 8 + j);
        int ci = K / 25, r25 = K - ci * 25;
        int ky = r25 / 5, kx = r25 - ky * 5;
        int idx = (ci * 20 + ly0 + ky) * 20 + col + kx;
        bool kv = ci < Cin;  // K-padding guard (only last kstep of Cin=3)
        b0[j] = kv ? sh[idx]      : (_Float16)0.f;
        b1[j] = kv ? sh[idx + 20] : (_Float16)0.f;
      }
      acc0 = __builtin_amdgcn_wmma_f32_16x16x32_f16(
          false, a, false, b0, (short)0, acc0, false, false);
      acc1 = __builtin_amdgcn_wmma_f32_16x16x32_f16(
          false, a, false, b1, (short)0, acc1, false, false);
    }
    // epilogue: C/D layout row M = r + 8*hi -> co; col N = l&15 -> pixel
#pragma unroll
    for (int rr = 0; rr < 2; ++rr) {
      int oy = oy0 + ly0 + rr;
      if (oy < Hout && ox < Wout) {
#pragma unroll
        for (int r = 0; r < 8; ++r) {
          int co = mt * 16 + 8 * hi + r;
          if (co < Cout) {
            float v = (rr ? acc1[r] : acc0[r]) + bias[co];
            if (doSelu) v = selu_f(v);
            int oidx = ((b * Cout + co) * Hout + oy) * Wout + ox;
            if (outh) outh[oidx] = (_Float16)v;
            else      outf[oidx] = v;
          }
        }
      }
    }
  }
}

// ------------------------------------------------------- VQ nearest-codebook
// argmin_n ||z - e_n||^2 == argmin_n (||e_n||^2 - 2 z.e_n).
__global__ void __launch_bounds__(256) k_vq_wmma(
    const _Float16* __restrict__ z, const _Float16* __restrict__ efrag,
    const float* __restrict__ e2, const float* __restrict__ emb,
    _Float16* __restrict__ q, int* __restrict__ idxout) {
  const int lane = threadIdx.x & 31;
  const int wave = threadIdx.x >> 5;
  const int col  = lane & 15;
  const int hi   = lane >> 4;
  const int HW   = 13456;  // 116*116; 32*HW divisible by 128 -> full tiles

  const int n0 = (blockIdx.x * 8 + wave) * 16;

  int p  = n0 + col;
  int b  = p / HW;
  int yx = p % HW;
  const _Float16* zp = z + (size_t)(b * 64) * HW + yx;
  v16h a0, a1;
#pragma unroll
  for (int j = 0; j < 16; ++j) {
    int k0 = hi * 16 + j;
    a0[j] = zp[(size_t)k0 * HW];
    a1[j] = zp[(size_t)(k0 + 32) * HW];
  }

  float best[8]; int bidx[8];
#pragma unroll
  for (int r = 0; r < 8; ++r) { best[r] = 3.4e38f; bidx[r] = 0; }

  for (int et = 0; et < 32; ++et) {
    v16h b0 = *(const v16h*)(efrag + ((size_t)et * 2 + 0) * 512 + lane * 16);
    v16h b1 = *(const v16h*)(efrag + ((size_t)et * 2 + 1) * 512 + lane * 16);
    v8f dac = {};
    dac = __builtin_amdgcn_wmma_f32_16x16x32_f16(
        false, a0, false, b0, (short)0, dac, false, false);
    dac = __builtin_amdgcn_wmma_f32_16x16x32_f16(
        false, a1, false, b1, (short)0, dac, false, false);
    int e = et * 16 + col;
    float ee = e2[e];
#pragma unroll
    for (int r = 0; r < 8; ++r) {
      float d = ee - 2.f * dac[r];
      if (d < best[r]) { best[r] = d; bidx[r] = e; }
    }
  }

#pragma unroll
  for (int r = 0; r < 8; ++r) {
#pragma unroll
    for (int m = 1; m < 16; m <<= 1) {
      float ob = __shfl_xor(best[r], m, 32);
      int   oi = __shfl_xor(bidx[r], m, 32);
      if (ob < best[r] || (ob == best[r] && oi < bidx[r])) {
        best[r] = ob; bidx[r] = oi;
      }
    }
  }

  // Pixels owned by this half-group: n0 + 8*hi + r; all 16 lanes agree.
#pragma unroll
  for (int r = 0; r < 8; ++r) {
    int pp  = n0 + 8 * hi + r;
    int bb  = pp / HW;
    int pyx = pp % HW;
    int e   = bidx[r];
#pragma unroll
    for (int t = 0; t < 4; ++t) {
      int d = col * 4 + t;  // 16 lanes x 4 dims = 64 dims
      q[((size_t)(bb * 64 + d)) * HW + pyx] = (_Float16)emb[e * 64 + d];
    }
    if (col == 0) idxout[pp] = e;
  }
}

// ---------------------------------------------------------------------- launch

extern "C" void kernel_launch(void* const* d_in, const int* in_sizes, int n_in,
                              void* d_out, int out_size, void* d_ws, size_t ws_size,
                              hipStream_t stream) {
  const float* x   = (const float*)d_in[0];
  const float* ew1 = (const float*)d_in[1];
  const float* eb1 = (const float*)d_in[2];
  const float* ew2 = (const float*)d_in[3];
  const float* eb2 = (const float*)d_in[4];
  const float* ew3 = (const float*)d_in[5];
  const float* eb3 = (const float*)d_in[6];
  const float* dw1 = (const float*)d_in[7];
  const float* db1 = (const float*)d_in[8];
  const float* dw2 = (const float*)d_in[9];
  const float* db2 = (const float*)d_in[10];
  const float* dw3 = (const float*)d_in[11];
  const float* db3 = (const float*)d_in[12];
  const float* emb = (const float*)d_in[13];

  float* recon = (float*)d_out;            // 32*3*128*128 = 1572864 f32
  int*   idxo  = (int*)(recon + 1572864);  // 32*116*116  = 430592 i32

  _Float16* w16 = (_Float16*)d_ws;
  size_t off = 0;
  auto take = [&](size_t n) { _Float16* p = w16 + off; off += n; return p; };
  _Float16* xh    = take(1572864);    // (32,3,128,128)
  _Float16* a1    = take(15745024);   // (32,32,124,124)
  _Float16* a2    = take(14745600);   // (32,32,120,120)
  _Float16* zb    = take(27557888);   // (32,64,116,116)
  _Float16* qb    = take(27557888);   // (32,64,116,116)
  _Float16* d1    = take(14745600);   // (32,32,120,120)
  _Float16* d2    = take(15745024);   // (32,32,124,124)
  _Float16* wf1   = take(3072);       // 2 mtiles * 3 ksteps  * 512
  _Float16* wf2   = take(25600);      // 2 * 25 * 512
  _Float16* wf3   = take(51200);      // 4 * 25 * 512
  _Float16* wf4   = take(51200);      // 2 * 50 * 512
  _Float16* wf5   = take(25600);      // 2 * 25 * 512
  _Float16* wf6   = take(12800);      // 1 * 25 * 512
  _Float16* efrag = take(32768);      // 32 etiles * 2 ksteps * 512
  float*    e2    = (float*)(w16 + off);  // 512 f32

  // ---- prep
  k_f32_to_f16<<<(1572864 + 255) / 256, 256, 0, stream>>>(x, xh, 1572864);
  k_pack_w<<<(3072  + 255) / 256, 256, 0, stream>>>(ew1, wf1, 32,  3,  3, 0, 3072);
  k_pack_w<<<(25600 + 255) / 256, 256, 0, stream>>>(ew2, wf2, 32, 32, 25, 0, 25600);
  k_pack_w<<<(51200 + 255) / 256, 256, 0, stream>>>(ew3, wf3, 64, 32, 25, 0, 51200);
  k_pack_w<<<(51200 + 255) / 256, 256, 0, stream>>>(dw1, wf4, 32, 64, 50, 1, 51200);
  k_pack_w<<<(25600 + 255) / 256, 256, 0, stream>>>(dw2, wf5, 32, 32, 25, 1, 25600);
  k_pack_w<<<(12800 + 255) / 256, 256, 0, stream>>>(dw3, wf6,  3, 32, 25, 1, 12800);
  k_pack_emb<<<32768 / 256, 256, 0, stream>>>(emb, efrag);
  k_emb_norm<<<2, 256, 0, stream>>>(emb, e2);

  // ---- all layers tile as 32 batch * 8x8 tiles of 16x16 = 2048 blocks
  // conv1: 3->32, 128->124
  k_conv_wmma<<<2048, 256, 3 * 400 * 2, stream>>>(
      xh, wf1, eb1, a1, nullptr, 3, 32, 128, 128, 124, 124, 0, 3, 2, 8, 8, 1);
  // conv2: 32->32, 124->120
  k_conv_wmma<<<2048, 256, 32 * 400 * 2, stream>>>(
      a1, wf2, eb2, a2, nullptr, 32, 32, 124, 124, 120, 120, 0, 25, 2, 8, 8, 1);
  // conv3: 32->64, 120->116 (no SELU on z)
  k_conv_wmma<<<2048, 256, 32 * 400 * 2, stream>>>(
      a2, wf3, eb3, zb, nullptr, 32, 64, 120, 120, 116, 116, 0, 25, 4, 8, 8, 0);

  // ---- vector quantize (WMMA distance GEMM + shuffle argmin)
  k_vq_wmma<<<430592 / 128, 256, 0, stream>>>(zb, efrag, e2, emb, qb, idxo);

  // ---- decoder (conv_transpose == pad-4 correlation, IOHW weights)
  k_conv_wmma<<<2048, 256, 64 * 400 * 2, stream>>>(
      qb, wf4, db1, d1, nullptr, 64, 32, 116, 116, 120, 120, 4, 50, 2, 8, 8, 1);
  k_conv_wmma<<<2048, 256, 32 * 400 * 2, stream>>>(
      d1, wf5, db2, d2, nullptr, 32, 32, 120, 120, 124, 124, 4, 25, 2, 8, 8, 1);
  // final: 32->3, f32 recon straight to d_out, no SELU
  k_conv_wmma<<<2048, 256, 32 * 400 * 2, stream>>>(
      d2, wf6, db3, nullptr, recon, 32, 3, 124, 124, 128, 128, 4, 25, 1, 8, 8, 0);
}
